// AttnDecoderLSTM_51616916963553
// MI455X (gfx1250) — matrix operand order repaired and checked by reading, not compiled, not run on hardware
//
#include <hip/hip_runtime.h>
#include <hip/hip_fp16.h>

// ---------------------------------------------------------------------------
// AttnDecoderLSTM single step, batch=1.  SEQ=2048, H=1024, VOCAB=32000.
// All stages are GEMVs -> HBM bandwidth bound (~332 MB => ~14us floor).
// The two big weight streams (W_ih/W_hh: 50MB, Wl: 262MB) are evaluated with
// v_wmma_f32_16x16x32_f16 (vector replicated in A, 16 weight rows as B
// columns, f32->f16 in-register, f32 accumulation).  Weight streams use
// non-temporal loads so the reused enc (8MB, read 2x) stays in the 192MB L2.
// ---------------------------------------------------------------------------

#define SEQ   2048
#define Hdim  1024
#define VOCAB 32000

typedef __attribute__((ext_vector_type(16))) _Float16 v16h;
typedef __attribute__((ext_vector_type(8)))  _Float16 v8h;
typedef __attribute__((ext_vector_type(8)))  float    v8f;
typedef __attribute__((ext_vector_type(4)))  float    v4f;

// ---------------- wave/block reductions (wave32) ----------------
__device__ inline float wave_sum(float v) {
#pragma unroll
  for (int o = 16; o > 0; o >>= 1) v += __shfl_down(v, o, 32);
  return v;
}
__device__ inline float wave_max(float v) {
#pragma unroll
  for (int o = 16; o > 0; o >>= 1) v = fmaxf(v, __shfl_down(v, o, 32));
  return v;
}
__device__ inline float block_sum(float v, float* sh) {
  const int lane = threadIdx.x & 31, w = threadIdx.x >> 5, nw = blockDim.x >> 5;
  v = wave_sum(v);
  if (lane == 0) sh[w] = v;
  __syncthreads();
  if (w == 0) {
    float x = (lane < nw) ? sh[lane] : 0.0f;
    x = wave_sum(x);
    if (lane == 0) sh[0] = x;
  }
  __syncthreads();
  float r = sh[0];
  __syncthreads();
  return r;
}
__device__ inline float block_max(float v, float* sh) {
  const int lane = threadIdx.x & 31, w = threadIdx.x >> 5, nw = blockDim.x >> 5;
  v = wave_max(v);
  if (lane == 0) sh[w] = v;
  __syncthreads();
  if (w == 0) {
    float x = (lane < nw) ? sh[lane] : -3.4e38f;
    x = wave_max(x);
    if (lane == 0) sh[0] = x;
  }
  __syncthreads();
  float r = sh[0];
  __syncthreads();
  return r;
}
__device__ inline float sigmoidf(float x) { return 1.0f / (1.0f + expf(-x)); }

// ---------------------------------------------------------------------------
// GEMV of 16 consecutive rows of a row-major f32 matrix against an f16 vector,
// executed by ONE wave over K range [k0, k0+klen), klen % 32 == 0.
//
// D = A x B with A[m][k] = v[k] (rows replicated) and B[k][n] = W[row16+n][k].
// A layout (16-bit A 16x32, contiguous k order as in B): lane<16 holds K = 0..15,
// lane>=16 holds K = 16..31.  B layout (32x16): lane n holds column
// n%16, K = hi*16 .. hi*16+15 as 16 packed halves.  All D rows are identical,
// so acc[0] of lane L is the dot product for weight row (row16 + L%16).
// Weights are streamed once -> non-temporal (protect L2 residency of enc).
// ---------------------------------------------------------------------------
__device__ inline float gemv16_wave(const float* __restrict__ W, int ld, int row16,
                                    const _Float16* __restrict__ vh, int k0, int klen) {
  const int lane = threadIdx.x & 31;
  const int col  = lane & 15;   // B column == weight row within the 16-row tile
  const int hi   = lane >> 4;   // lane half selects K sub-range
  const float* wrow = W + (size_t)(row16 + col) * (size_t)ld;
  v8f acc = {};
  for (int k = k0; k < k0 + klen; k += 32) {
    const v4f* wp = (const v4f*)(wrow + k + hi * 16);
    __builtin_prefetch(wrow + k + 256, 0, 0);          // global_prefetch_b8
    v4f w0 = __builtin_nontemporal_load(wp + 0);       // TH_NT weight stream
    v4f w1 = __builtin_nontemporal_load(wp + 1);
    v4f w2 = __builtin_nontemporal_load(wp + 2);
    v4f w3 = __builtin_nontemporal_load(wp + 3);
    v16h b;
#pragma unroll
    for (int j = 0; j < 4; ++j) {
      b[j]      = (_Float16)w0[j];
      b[4 + j]  = (_Float16)w1[j];
      b[8 + j]  = (_Float16)w2[j];
      b[12 + j] = (_Float16)w3[j];
    }
    const v8h a0 = *(const v8h*)(vh + k + hi * 16);
    const v8h a1 = *(const v8h*)(vh + k + hi * 16 + 8);
    v16h a;
#pragma unroll
    for (int j = 0; j < 8; ++j) { a[j] = a0[j]; a[8 + j] = a1[j]; }
    acc = __builtin_amdgcn_wmma_f32_16x16x32_f16(
        /*neg_a=*/false, a, /*neg_b=*/false, b,
        /*c_mod=*/(short)0, acc, /*reuse_a=*/false, /*reuse_b=*/false);
  }
  return acc[0];
}

// ---------------- kernel 1: embedding lookup + f16 input vectors -----------
__global__ void prep_kernel(const int* __restrict__ word,
                            const float* __restrict__ emb,
                            const float* __restrict__ last_ctx,
                            const float* __restrict__ last_h,
                            __half* __restrict__ vin_h,   // [2H] = [emb(idx); last_ctx]
                            __half* __restrict__ h0_h) {  // [H]
  const int t = threadIdx.x;
  const size_t idx = (size_t)word[0];
  vin_h[t]        = __float2half(emb[idx * Hdim + t]);
  vin_h[Hdim + t] = __float2half(last_ctx[t]);
  h0_h[t]         = __float2half(last_h[t]);
}

// ---------------- kernel 2: gates = W_ih@vin + b_ih + W_hh@h0 + b_hh -------
// 256 blocks x 128 threads; block -> 16 gate rows; waves 0-1 split W_ih's
// K=2048, waves 2-3 split W_hh's K=1024.  LDS reduce.
__global__ void gates_kernel(const float* __restrict__ W_ih,
                             const float* __restrict__ W_hh,
                             const float* __restrict__ b_ih,
                             const float* __restrict__ b_hh,
                             const __half* __restrict__ vin_h,
                             const __half* __restrict__ h0_h,
                             float* __restrict__ gates) {
  __shared__ float part[64];
  const int wave = threadIdx.x >> 5, lane = threadIdx.x & 31;
  const int row16 = blockIdx.x * 16;
  float r;
  if (wave < 2)
    r = gemv16_wave(W_ih, 2 * Hdim, row16,
                    (const _Float16*)vin_h, wave * 1024, 1024);
  else
    r = gemv16_wave(W_hh, Hdim, row16,
                    (const _Float16*)h0_h, (wave - 2) * 512, 512);
  if (lane < 16) part[wave * 16 + lane] = r;
  __syncthreads();
  if (threadIdx.x < 16) {
    const int row = row16 + threadIdx.x;
    gates[row] = part[threadIdx.x] + part[16 + threadIdx.x] +
                 part[32 + threadIdx.x] + part[48 + threadIdx.x] +
                 b_ih[row] + b_hh[row];
  }
}

// ---------------- kernel 3: LSTM activations (c0 == 0) ---------------------
__global__ void lstm_act_kernel(const float* __restrict__ gates,
                                float* __restrict__ h) {
  const int t = threadIdx.x;
  const float i_g = gates[t];
  const float g_g = gates[2 * Hdim + t];
  const float o_g = gates[3 * Hdim + t];
  const float c = sigmoidf(i_g) * tanhf(g_g);   // f-gate * c0 == 0
  h[t] = sigmoidf(o_g) * tanhf(c);
}

// ---------------- kernel 4: u = Wa^T @ h  (column-GEMV, h staged in LDS) ---
// (ba drops out: it only shifts every energy by ba.h, softmax-invariant.)
__global__ void wa_kernel(const float* __restrict__ Wa,
                          const float* __restrict__ h,
                          float* __restrict__ u) {
  __shared__ float hs[Hdim];
  const int t = threadIdx.x;
  for (int i = t; i < Hdim; i += 256) hs[i] = h[i];
  __syncthreads();
  const int j = blockIdx.x * 256 + t;
  float acc = 0.0f;
#pragma unroll 4
  for (int i = 0; i < Hdim; ++i) acc += Wa[(size_t)i * Hdim + j] * hs[i];
  u[j] = acc;
}

// ---------------- kernel 5: energies_i = enc_i . u  (wave per row) ---------
__global__ void energy_kernel(const float* __restrict__ enc,
                              const float* __restrict__ u,
                              float* __restrict__ e) {
  const int wave = threadIdx.x >> 5, lane = threadIdx.x & 31;
  const int row = blockIdx.x * 8 + wave;
  const float4* er = (const float4*)(enc + (size_t)row * Hdim);
  const float4* uu = (const float4*)u;
  float acc = 0.0f;
  for (int i = lane; i < Hdim / 4; i += 32) {
    float4 a = er[i], b = uu[i];
    acc += a.x * b.x + a.y * b.y + a.z * b.z + a.w * b.w;
  }
  acc = wave_sum(acc);
  if (lane == 0) e[row] = acc;
}

// ---------------- kernel 6: softmax over SEQ=2048 (one block) --------------
__global__ void softmax_kernel(float* __restrict__ e) {
  __shared__ float sh[32];
  const int t = threadIdx.x;
  const float e0 = e[t], e1 = e[Hdim + t];
  const float m = block_max(fmaxf(e0, e1), sh);
  const float x0 = expf(e0 - m), x1 = expf(e1 - m);
  const float s = block_sum(x0 + x1, sh);
  const float inv = 1.0f / s;
  e[t] = x0 * inv;
  e[Hdim + t] = x1 * inv;
}

// ---------------- kernel 7: context_j = sum_i w_i enc[i][j]; pack [h;ctx] --
// Attention weights staged into LDS with gfx1250 async DMA (ASYNCcnt).
__global__ void context_kernel(const float* __restrict__ enc,
                               const float* __restrict__ w,
                               const float* __restrict__ h,
                               float* __restrict__ ctx,
                               __half* __restrict__ hc_h) {
  __shared__ float ws_[SEQ];
  const int t = threadIdx.x;
  // global_load_async_to_lds_b128: VDST = LDS byte offset (low 32 bits of the
  // generic pointer to the __shared__ element), VADDR = global address.
#pragma unroll
  for (int i = t * 4; i < SEQ; i += 256 * 4) {
    unsigned lds_off = (unsigned)(size_t)&ws_[i];
    const float* g = w + i;
    asm volatile("global_load_async_to_lds_b128 %0, %1, off"
                 :: "v"(lds_off), "v"(g) : "memory");
  }
  asm volatile("s_wait_asynccnt 0x0" ::: "memory");
  __syncthreads();
  const int j = blockIdx.x * 256 + t;
  float acc = 0.0f;
#pragma unroll 4
  for (int i = 0; i < SEQ; ++i) acc += ws_[i] * enc[(size_t)i * Hdim + j];
  ctx[j] = acc;
  hc_h[j]        = __float2half(h[j]);
  hc_h[Hdim + j] = __float2half(acc);
}

// ---------------- kernel 8: logits = Wl @ [h;ctx] + bl (the 262MB stream) --
// 2000 blocks x 128 threads; 16 vocab rows per block; 4 waves split K=2048.
__global__ void logits_kernel(const float* __restrict__ Wl,
                              const float* __restrict__ bl,
                              const __half* __restrict__ hc_h,
                              float* __restrict__ logits) {
  __shared__ float part[64];
  const int wave = threadIdx.x >> 5, lane = threadIdx.x & 31;
  const int row16 = blockIdx.x * 16;
  float r = gemv16_wave(Wl, 2 * Hdim, row16,
                        (const _Float16*)hc_h, wave * 512, 512);
  if (lane < 16) part[wave * 16 + lane] = r;
  __syncthreads();
  if (threadIdx.x < 16) {
    const int row = row16 + threadIdx.x;
    logits[row] = part[threadIdx.x] + part[16 + threadIdx.x] +
                  part[32 + threadIdx.x] + part[48 + threadIdx.x] + bl[row];
  }
}

// ---------------- kernel 9a: per-block partial (max, sum-exp) over VOCAB ---
__global__ void lse_part_kernel(const float* __restrict__ logits,
                                float* __restrict__ pm,
                                float* __restrict__ ps) {
  __shared__ float sh[32];
  const int i = blockIdx.x * 256 + threadIdx.x;
  const float x = logits[i];
  const float m = block_max(x, sh);
  const float s = block_sum(expf(x - m), sh);
  if (threadIdx.x == 0) { pm[blockIdx.x] = m; ps[blockIdx.x] = s; }
}

// ---------------- kernel 9b: combine partials: M, log(S) -------------------
__global__ void lse_combine_kernel(const float* __restrict__ pm,
                                   const float* __restrict__ ps,
                                   int nblk, float* __restrict__ red) {
  __shared__ float sh[32];
  const int t = threadIdx.x;
  const float mb = (t < nblk) ? pm[t] : -3.4e38f;
  const float sb = (t < nblk) ? ps[t] : 0.0f;
  const float M = block_max(mb, sh);
  const float S = block_sum(sb * expf(mb - M), sh);
  if (t == 0) { red[0] = M; red[1] = logf(S); }
}

// ---------------- kernel 10: out = logits - M - log(S) ---------------------
__global__ void out_kernel(const float* __restrict__ logits,
                           const float* __restrict__ red,
                           float* __restrict__ out) {
  const int v = blockIdx.x * 256 + threadIdx.x;
  out[v] = logits[v] - red[0] - red[1];
}

// ---------------------------------------------------------------------------
extern "C" void kernel_launch(void* const* d_in, const int* in_sizes, int n_in,
                              void* d_out, int out_size, void* d_ws, size_t ws_size,
                              hipStream_t stream) {
  (void)in_sizes; (void)n_in; (void)out_size; (void)ws_size;
  const int*   word    = (const int*)  d_in[0];
  const float* lastctx = (const float*)d_in[1];
  const float* lasth   = (const float*)d_in[2];
  const float* enc     = (const float*)d_in[3];
  const float* emb     = (const float*)d_in[4];
  const float* W_ih    = (const float*)d_in[5];
  const float* W_hh    = (const float*)d_in[6];
  const float* b_ih    = (const float*)d_in[7];
  const float* b_hh    = (const float*)d_in[8];
  const float* Wa      = (const float*)d_in[9];
  // d_in[10] = ba : provably unused (softmax shift invariance)
  const float* Wl      = (const float*)d_in[11];
  const float* bl      = (const float*)d_in[12];

  char* w8 = (char*)d_ws;                     // scratch layout (~177 KB)
  float*  logits = (float*)(w8 + 0);          // 32000 f32
  float*  gates  = (float*)(w8 + 128000);     //  4096 f32
  float*  hbuf   = (float*)(w8 + 144384);     //  1024 f32
  float*  u      = (float*)(w8 + 148480);     //  1024 f32
  float*  eattn  = (float*)(w8 + 152576);     //  2048 f32
  float*  ctx    = (float*)(w8 + 160768);     //  1024 f32
  __half* vin_h  = (__half*)(w8 + 164864);    //  2048 f16
  __half* h0_h   = (__half*)(w8 + 168960);    //  1024 f16
  __half* hc_h   = (__half*)(w8 + 171008);    //  2048 f16
  float*  red    = (float*)(w8 + 175104);     //     2 f32
  float*  pm     = (float*)(w8 + 175136);     //   125 f32
  float*  ps     = (float*)(w8 + 175648);     //   125 f32

  prep_kernel       <<<dim3(1),    dim3(1024), 0, stream>>>(word, emb, lastctx, lasth, vin_h, h0_h);
  gates_kernel      <<<dim3(256),  dim3(128),  0, stream>>>(W_ih, W_hh, b_ih, b_hh, vin_h, h0_h, gates);
  lstm_act_kernel   <<<dim3(1),    dim3(1024), 0, stream>>>(gates, hbuf);
  wa_kernel         <<<dim3(4),    dim3(256),  0, stream>>>(Wa, hbuf, u);
  energy_kernel     <<<dim3(256),  dim3(256),  0, stream>>>(enc, u, eattn);
  softmax_kernel    <<<dim3(1),    dim3(1024), 0, stream>>>(eattn);
  context_kernel    <<<dim3(4),    dim3(256),  0, stream>>>(enc, eattn, hbuf, ctx, hc_h);
  logits_kernel     <<<dim3(2000), dim3(128),  0, stream>>>(Wl, bl, hc_h, logits);
  lse_part_kernel   <<<dim3(125),  dim3(256),  0, stream>>>(logits, pm, ps);
  lse_combine_kernel<<<dim3(1),    dim3(128),  0, stream>>>(pm, ps, 125, red);
  out_kernel        <<<dim3(125),  dim3(256),  0, stream>>>(logits, red, (float*)d_out);
}
